// DotGatConv_33311766348130
// MI455X (gfx1250) — matrix operand order repaired and checked by reading, not compiled, not run on hardware
//
#include <hip/hip_runtime.h>
#include <math.h>

#define N_NODES 100000
#define N_EDGES 1600000
#define IN_F    128
#define OUT_F   64
#define M_TILES (N_NODES / 16)   // 6250, exact

typedef float  v2f  __attribute__((ext_vector_type(2)));
typedef float  v8f  __attribute__((ext_vector_type(8)));
typedef __bf16 v16bf __attribute__((ext_vector_type(16)));

// ---------------------------------------------------------------------------
// Kernel 1: ft = feat @ W  via V_WMMA_F32_16X16X4_F32 (fp32-exact, mem-bound)
// One wave computes a 16x64 output row-block: 4 N-tiles of 16x16, K-loop of 32.
// A 16x4 f32 layout: lane<16 -> M=lane, K={0,1}; lane>=16 -> same M, K={2,3}.
// C/D 16x16 f32: VGPR r holds M=r (lanes 0-15) / M=r+8 (lanes 16-31), N=lane&15.
// ---------------------------------------------------------------------------
__global__ __launch_bounds__(256) void gemm_wmma(const float* __restrict__ feat,
                                                 const float* __restrict__ W,
                                                 float* __restrict__ ft) {
    const int lane   = threadIdx.x & 31;
    const int wave   = threadIdx.x >> 5;
    const int tile_m = blockIdx.x * 8 + wave;
    if (tile_m >= M_TILES) return;          // uniform per wave -> EXEC all-1s

    const int row  = lane & 15;
    const int half = lane >> 4;             // 0 or 1
    const float* Abase = feat + (size_t)(tile_m * 16 + row) * IN_F;

    v8f c0 = {}, c1 = {}, c2 = {}, c3 = {};

#if __has_builtin(__builtin_amdgcn_wmma_f32_16x16x4_f32)
    for (int k0 = 0; k0 < IN_F; k0 += 4) {
        const int ka = k0 + half * 2;
        v2f a;
        a.x = Abase[ka];
        a.y = Abase[ka + 1];
        const float* Bb = W + (size_t)ka * OUT_F + row;   // col = ntile*16 + row
        v2f b0, b1, b2, b3;
        b0.x = Bb[0];  b0.y = Bb[OUT_F + 0];
        b1.x = Bb[16]; b1.y = Bb[OUT_F + 16];
        b2.x = Bb[32]; b2.y = Bb[OUT_F + 32];
        b3.x = Bb[48]; b3.y = Bb[OUT_F + 48];
        c0 = __builtin_amdgcn_wmma_f32_16x16x4_f32(false, a, false, b0, (short)0, c0, false, false);
        c1 = __builtin_amdgcn_wmma_f32_16x16x4_f32(false, a, false, b1, (short)0, c1, false, false);
        c2 = __builtin_amdgcn_wmma_f32_16x16x4_f32(false, a, false, b2, (short)0, c2, false, false);
        c3 = __builtin_amdgcn_wmma_f32_16x16x4_f32(false, a, false, b3, (short)0, c3, false, false);
    }
#else
    // bf16 WMMA fallback (codegen-confirmed builtin), K-step 32.
    for (int k0 = 0; k0 < IN_F; k0 += 32) {
        v16bf a;
        #pragma unroll
        for (int j = 0; j < 16; ++j) {
            const int K = k0 + half * 16 + j;
            a[j] = (__bf16)Abase[K];
        }
        v16bf b0, b1, b2, b3;
        #pragma unroll
        for (int j = 0; j < 16; ++j) {
            const int K = k0 + half * 16 + j;
            b0[j] = (__bf16)W[(size_t)K * OUT_F + row +  0];
            b1[j] = (__bf16)W[(size_t)K * OUT_F + row + 16];
            b2[j] = (__bf16)W[(size_t)K * OUT_F + row + 32];
            b3[j] = (__bf16)W[(size_t)K * OUT_F + row + 48];
        }
        c0 = __builtin_amdgcn_wmma_f32_16x16x32_bf16(false, a, false, b0, (short)0, c0, false, false);
        c1 = __builtin_amdgcn_wmma_f32_16x16x32_bf16(false, a, false, b1, (short)0, c1, false, false);
        c2 = __builtin_amdgcn_wmma_f32_16x16x32_bf16(false, a, false, b2, (short)0, c2, false, false);
        c3 = __builtin_amdgcn_wmma_f32_16x16x32_bf16(false, a, false, b3, (short)0, c3, false, false);
    }
#endif

    float* O = ft + (size_t)(tile_m * 16 + half * 8) * OUT_F + row;
    #pragma unroll
    for (int r = 0; r < 8; ++r) {
        O[(size_t)r * OUT_F +  0] = c0[r];
        O[(size_t)r * OUT_F + 16] = c1[r];
        O[(size_t)r * OUT_F + 32] = c2[r];
        O[(size_t)r * OUT_F + 48] = c3[r];
    }
}

// Monotone float -> uint key (unsigned max ordering == float ordering)
__device__ __forceinline__ unsigned f2key(float f) {
    unsigned b = __float_as_uint(f);
    return (b & 0x80000000u) ? ~b : (b | 0x80000000u);
}

// ---------------------------------------------------------------------------
// Kernel 2: per-edge dot score + segmented max (one wave32 per edge)
// ---------------------------------------------------------------------------
__global__ __launch_bounds__(256) void edge_score(const float* __restrict__ ft,
                                                  const int* __restrict__ src,
                                                  const int* __restrict__ dst,
                                                  float* __restrict__ a,
                                                  unsigned* __restrict__ mkeys) {
    const int e    = (int)((blockIdx.x * (unsigned)blockDim.x + threadIdx.x) >> 5);
    const int lane = threadIdx.x & 31;
    if (e >= N_EDGES) return;
    const int s = src[e], d = dst[e];
    const float2 x = ((const float2*)(ft + (size_t)s * OUT_F))[lane];
    const float2 y = ((const float2*)(ft + (size_t)d * OUT_F))[lane];
    float p = x.x * y.x + x.y * y.y;
    #pragma unroll
    for (int off = 16; off > 0; off >>= 1) p += __shfl_xor(p, off, 32);
    if (lane == 0) {
        a[e] = p;
        atomicMax(&mkeys[d], f2key(p));
    }
}

// ---------------------------------------------------------------------------
// Kernel 3: invert keys -> float max, guard zero-in-degree (key 0 -> m = 0)
// ---------------------------------------------------------------------------
__global__ __launch_bounds__(256) void unmap_max(unsigned* __restrict__ mkeys) {
    const int i = blockIdx.x * blockDim.x + threadIdx.x;
    if (i >= N_NODES) return;
    const unsigned k = mkeys[i];
    float m;
    if (k == 0u)                 m = 0.0f;
    else if (k & 0x80000000u)    m = __uint_as_float(k ^ 0x80000000u);
    else                         m = __uint_as_float(~k);
    ((float*)mkeys)[i] = m;
}

// ---------------------------------------------------------------------------
// Kernel 4: e = exp(a - m[dst]); segmented sum (thread per edge)
// ---------------------------------------------------------------------------
__global__ __launch_bounds__(256) void edge_exp(const float* __restrict__ m,
                                                const int* __restrict__ dst,
                                                float* __restrict__ a,
                                                float* __restrict__ ssum) {
    const int e = blockIdx.x * blockDim.x + threadIdx.x;
    if (e >= N_EDGES) return;
    const int d = dst[e];
    const float ev = expf(a[e] - m[d]);
    a[e] = ev;
    atomicAdd(&ssum[d], ev);
}

// ---------------------------------------------------------------------------
// Kernel 5: out[dst] += (e/s[dst]) * ft[src]  (one wave32 per edge)
// ---------------------------------------------------------------------------
__global__ __launch_bounds__(256) void edge_scatter(const float* __restrict__ ft,
                                                    const int* __restrict__ src,
                                                    const int* __restrict__ dst,
                                                    const float* __restrict__ a,
                                                    const float* __restrict__ ssum,
                                                    float* __restrict__ out) {
    const int e    = (int)((blockIdx.x * (unsigned)blockDim.x + threadIdx.x) >> 5);
    const int lane = threadIdx.x & 31;
    if (e >= N_EDGES) return;
    const int s = src[e], d = dst[e];
    const float coef = a[e] / ssum[d];
    const float2 x = ((const float2*)(ft + (size_t)s * OUT_F))[lane];
    float* o = out + (size_t)d * OUT_F + lane * 2;
    atomicAdd(o,     coef * x.x);
    atomicAdd(o + 1, coef * x.y);
}

extern "C" void kernel_launch(void* const* d_in, const int* in_sizes, int n_in,
                              void* d_out, int out_size, void* d_ws, size_t ws_size,
                              hipStream_t stream) {
    const float* feat = (const float*)d_in[0];
    const float* W    = (const float*)d_in[1];
    const int*   src  = (const int*)d_in[2];
    const int*   dst  = (const int*)d_in[3];
    float* out = (float*)d_out;

    // Workspace layout (≈ 33 MB): ft | a | mkeys/m | ssum
    float*    ft    = (float*)d_ws;
    float*    a     = ft + (size_t)N_NODES * OUT_F;
    unsigned* mkeys = (unsigned*)(a + N_EDGES);
    float*    ssum  = (float*)(mkeys + N_NODES);

    hipMemsetAsync(out,   0, (size_t)N_NODES * OUT_F * sizeof(float), stream);
    hipMemsetAsync(mkeys, 0, (size_t)N_NODES * sizeof(unsigned), stream);
    hipMemsetAsync(ssum,  0, (size_t)N_NODES * sizeof(float), stream);

    gemm_wmma<<<(M_TILES + 7) / 8, 256, 0, stream>>>(feat, W, ft);
    edge_score<<<(N_EDGES + 7) / 8, 256, 0, stream>>>(ft, src, dst, a, mkeys);
    unmap_max<<<(N_NODES + 255) / 256, 256, 0, stream>>>(mkeys);
    edge_exp<<<(N_EDGES + 255) / 256, 256, 0, stream>>>((const float*)mkeys, dst, a, ssum);
    edge_scatter<<<(N_EDGES + 7) / 8, 256, 0, stream>>>(ft, src, dst, a, ssum, out);
}
